// BiDAF_62225486185128
// MI455X (gfx1250) — matrix operand order repaired and checked by reading, not compiled, not run on hardware
//
#include <hip/hip_runtime.h>
#include <hip/hip_bf16.h>

// ---------------- model dims ----------------
#define BSZ   16
#define CLENL 400
#define QLENL 48
#define EMBD  300
#define EMBP  320          // EMB padded to a multiple of 32 (WMMA K granularity)
#define HD    128          // hidden
#define H2    256          // 2H
#define H4    512          // 4H (i,f,g,o gates)
#define H8    1024         // 8H
#define MC    (BSZ*CLENL)  // 6400 context rows
#define MQ    (BSZ*QLENL)  // 768 question rows

typedef __attribute__((ext_vector_type(16))) _Float16 v16h;
typedef __attribute__((ext_vector_type(8)))  _Float16 v8h;
typedef __attribute__((ext_vector_type(8)))  float    v8f;
typedef __attribute__((ext_vector_type(4)))  unsigned int u32x4;
typedef __attribute__((ext_vector_type(8)))  int      i32x8;
typedef __attribute__((ext_vector_type(4)))  int      i32x4;

static __device__ __forceinline__ v16h cat8(v8h lo, v8h hi) {
  return __builtin_shufflevector(lo, hi, 0,1,2,3,4,5,6,7,8,9,10,11,12,13,14,15);
}
static __device__ __forceinline__ float sigmoidf_(float x) {
  return 1.0f / (1.0f + __expf(-x));
}

// ---------------- f32 -> f16 weight conversion with K-padding ----------------
__global__ void cvt_pad(const float* __restrict__ src, _Float16* __restrict__ dst,
                        int rows, int k, int kp) {
  int i = blockIdx.x * blockDim.x + threadIdx.x;
  if (i >= rows * kp) return;
  int row = i / kp, j = i - row * kp;
  dst[i] = (j < k) ? (_Float16)src[(size_t)row * k + j] : (_Float16)0.0f;
}

// ---------------- embedding gather -> padded f16 rows ----------------
__global__ void embed_kernel(const float* __restrict__ emb, const int* __restrict__ tok,
                             _Float16* __restrict__ dst, int rows) {
  int i = blockIdx.x * blockDim.x + threadIdx.x;
  if (i >= rows * EMBP) return;
  int row = i / EMBP, j = i - row * EMBP;
  float v = (j < EMBD) ? emb[(size_t)tok[row] * EMBD + j] : 0.0f;
  dst[i] = (_Float16)v;
}

// ---------------- WMMA GEMM: C[MxN] = A[MxK] * Bw[NxK]^T + bias[N] ----------------
// One wave32 per 16x64 output stripe (4 N-tiles): the A fragment is loaded once per
// K-chunk and reused by 4 WMMAs (4x arithmetic intensity on the activation stream).
// f32 accumulate via v_wmma_f32_16x16x32_f16. M mult of 16, N mult of 64, K mult of 32.
__global__ __launch_bounds__(32) void wmma_gemm_bias(
    const _Float16* __restrict__ A, const _Float16* __restrict__ Bw,
    const float* __restrict__ bias, float* __restrict__ C,
    int M, int N, int K) {
  const int tn0 = blockIdx.x * 64;
  const int tm  = blockIdx.y * 16;
  const int lane = threadIdx.x;
  const int r  = lane & 15;
  const int hi = lane >> 4;
  const _Float16* ap = A + (size_t)(tm + r) * K + hi * 8;   // A frag: split-K layout
  v8f acc[4] = {};
  for (int kc = 0; kc < K; kc += 32) {
    v16h af = cat8(*(const v8h*)(ap + kc), *(const v8h*)(ap + kc + 16));
#pragma unroll
    for (int qq = 0; qq < 4; ++qq) {
      const _Float16* bp = Bw + (size_t)(tn0 + qq * 16 + r) * K + hi * 8; // split-K layout
      v16h bf = cat8(*(const v8h*)(bp + kc), *(const v8h*)(bp + kc + 16));
      acc[qq] = __builtin_amdgcn_wmma_f32_16x16x32_f16(false, af, false, bf,
                                                       (short)0, acc[qq], false, false);
    }
  }
#pragma unroll
  for (int qq = 0; qq < 4; ++qq) {
    const int n = tn0 + qq * 16 + r;
    const float bv = bias ? bias[n] : 0.0f;
#pragma unroll
    for (int i = 0; i < 8; ++i)
      C[(size_t)(tm + hi * 8 + i) * N + n] = acc[qq][i] + bv;
  }
}

// ---------------- LSTM recurrence (one workgroup per direction) ----------------
// gates(t) = Xg[t] (precomputed x@Wih^T + b) + h(t-1) @ Whh^T.
// Whh (512x128 f16, 128 KB) is DMA'd into LDS once by the Tensor Data Mover, so the
// 400 sequential steps read B-fragments from LDS only. h/c state also lives in LDS.
// Dynamic LDS layout (base offset 0 == D#.lds_addr):
#define LDS_WHH_BYTES (H4 * HD * 2)                 // 131072
#define LDS_H_OFF     (LDS_WHH_BYTES)               // f16 h state, 4 KB
#define LDS_C_OFF     (LDS_H_OFF + BSZ * HD * 2)    // f32 c state, 8 KB
#define LDS_G_OFF     (LDS_C_OFF + BSZ * HD * 4)    // f32 gates, 32 KB
#define LDS_TOTAL     (LDS_G_OFF + BSZ * H4 * 4)    // 176128 bytes

__global__ __launch_bounds__(256) void lstm_recur(
    const float* __restrict__ xgf, const float* __restrict__ xgb,
    const _Float16* __restrict__ whhf, const _Float16* __restrict__ whhb,
    float* __restrict__ out, _Float16* __restrict__ outh, int T) {
  extern __shared__ __align__(16) char smem[];
  _Float16* whs    = (_Float16*)smem;
  _Float16* hprev  = (_Float16*)(smem + LDS_H_OFF);
  float*    cstate = (float*)(smem + LDS_C_OFF);
  float*    Gs     = (float*)(smem + LDS_G_OFF);

  const int dir = blockIdx.x;
  const float*    xg  = dir ? xgb  : xgf;
  const _Float16* whh = dir ? whhb : whhf;
  const int tid = threadIdx.x;

  // --- TDM: DMA Whh (2-D tensor, 512 rows x 128 f16, stride 128) -> LDS offset 0 ---
  if (tid < 32) {
    const unsigned long long ga = (unsigned long long)(uintptr_t)whh;
    u32x4 g0 = { 1u,                                    // count=1, no gather
                 0u,                                    // lds_addr = 0 (dynamic LDS base)
                 (unsigned)ga,                          // global_addr[31:0]
                 ((unsigned)(ga >> 32) & 0x1FFFFFFu) | 0x80000000u }; // addr[56:32] | type=2
    i32x8 g1 = { (int)(1u << 16),                       // data_size = 2 bytes
                 (int)(HD << 16),                       // tensor_dim0 = 128
                 (int)(H4 << 16),                       // tensor_dim1 = 512
                 (int)(HD << 16),                       // tile_dim0 = 128
                 (int)H4,                               // tile_dim1 = 512, tile_dim2 = 0
                 (int)HD,                               // tensor_dim0_stride = 128
                 0, 0 };
    i32x4 gz  = { 0, 0, 0, 0 };
    i32x8 gz8 = { 0, 0, 0, 0, 0, 0, 0, 0 };
    __builtin_amdgcn_tensor_load_to_lds(g0, g1, gz, gz, gz8, 0);
    __builtin_amdgcn_s_wait_tensorcnt(0);
  }
  for (int u = tid; u < BSZ * HD; u += 256) {
    hprev[u]  = (_Float16)0.0f;
    cstate[u] = 0.0f;
  }
  __syncthreads();

  const int lane = tid & 31, wave = tid >> 5;
  const int r = lane & 15, hi = lane >> 4;
  const int tn0 = wave * 64;   // this wave's 4 N-tiles: tn0 .. tn0+63

  for (int s = 0; s < T; ++s) {
    const int t = dir ? (T - 1 - s) : s;
    v8f acc[4] = {};
    for (int kc = 0; kc < HD; kc += 32) {
      v16h af = cat8(*(const v8h*)(hprev + r * HD + kc + hi * 8),
                     *(const v8h*)(hprev + r * HD + kc + 16 + hi * 8));
#pragma unroll
      for (int qq = 0; qq < 4; ++qq) {
        const _Float16* bp = whs + (size_t)(tn0 + qq * 16 + r) * HD + kc + hi * 8;
        v16h bf = cat8(*(const v8h*)bp, *(const v8h*)(bp + 16));
        acc[qq] = __builtin_amdgcn_wmma_f32_16x16x32_f16(false, af, false, bf,
                                                         (short)0, acc[qq], false, false);
      }
    }
#pragma unroll
    for (int qq = 0; qq < 4; ++qq) {
      const int n = tn0 + qq * 16 + r;
#pragma unroll
      for (int i = 0; i < 8; ++i) {
        const int bb = hi * 8 + i;
        Gs[bb * H4 + n] = acc[qq][i] + xg[((size_t)bb * T + t) * H4 + n];
      }
    }
    __syncthreads();
    for (int u = tid; u < BSZ * HD; u += 256) {
      const int bb = u >> 7, h = u & (HD - 1);
      const float ig = Gs[bb * H4 + h];
      const float fg = Gs[bb * H4 + HD + h];
      const float gg = Gs[bb * H4 + 2 * HD + h];
      const float og = Gs[bb * H4 + 3 * HD + h];
      const float cc = sigmoidf_(fg) * cstate[u] + sigmoidf_(ig) * tanhf(gg);
      const float hn = sigmoidf_(og) * tanhf(cc);
      cstate[u] = cc;
      hprev[u]  = (_Float16)hn;
      const size_t o = ((size_t)bb * T + t) * H2 + dir * HD + h;
      out[o] = hn;
      if (outh) outh[o] = (_Float16)hn;
    }
    __syncthreads();
  }
}

// ---------------- rowwise dot with a 2H vector ----------------
__global__ void dotw_kernel(const float* __restrict__ X, const float* __restrict__ w,
                            float* __restrict__ out, int n) {
  int i = blockIdx.x * blockDim.x + threadIdx.x;
  if (i >= n) return;
  const float* xr = X + (size_t)i * H2;
  float s = 0.0f;
  for (int j = 0; j < H2; ++j) s += xr[j] * w[j];
  out[i] = s;
}

// ---------------- c2q attention: per (b,i) row softmax over Q, attended q ----------------
__global__ __launch_bounds__(64) void attn_kernel(
    const float* __restrict__ c, const float* __restrict__ q,
    const float* __restrict__ cdot, const float* __restrict__ qdot,
    const float* __restrict__ wcq, float* __restrict__ aq, float* __restrict__ Smax) {
  const int m = blockIdx.x;        // b*CLENL + i
  const int b = m / CLENL;
  const int tid = threadIdx.x;
  __shared__ float cw[H2];
  __shared__ float sv[QLENL];
  for (int d = tid; d < H2; d += 64) cw[d] = c[(size_t)m * H2 + d] * wcq[d];
  __syncthreads();
  if (tid < QLENL) {
    const float* qr = q + ((size_t)b * QLENL + tid) * H2;
    float s = cdot[m] + qdot[b * QLENL + tid];
    for (int d = 0; d < H2; ++d) s += cw[d] * qr[d];
    sv[tid] = s;
  }
  __syncthreads();
  if (tid == 0) {
    float mx = sv[0];
    for (int j = 1; j < QLENL; ++j) mx = fmaxf(mx, sv[j]);
    Smax[m] = mx;                  // raw row max, feeds q2c softmax
    float sum = 0.0f;
    for (int j = 0; j < QLENL; ++j) { float e = __expf(sv[j] - mx); sv[j] = e; sum += e; }
    float inv = 1.0f / sum;
    for (int j = 0; j < QLENL; ++j) sv[j] *= inv;
  }
  __syncthreads();
  for (int d = tid; d < H2; d += 64) {
    float acc = 0.0f;
    for (int j = 0; j < QLENL; ++j) acc += sv[j] * q[((size_t)b * QLENL + j) * H2 + d];
    aq[(size_t)m * H2 + d] = acc;
  }
}

// ---------------- q2c attention: softmax over C of Smax, attended c ----------------
__global__ __launch_bounds__(256) void q2c_kernel(
    const float* __restrict__ Smax, const float* __restrict__ c, float* __restrict__ ac) {
  const int b = blockIdx.x, tid = threadIdx.x;
  __shared__ float bw[CLENL];
  __shared__ float red[256];
  float mx = -3.4e38f;
  for (int i = tid; i < CLENL; i += 256) mx = fmaxf(mx, Smax[b * CLENL + i]);
  red[tid] = mx; __syncthreads();
  for (int st = 128; st > 0; st >>= 1) { if (tid < st) red[tid] = fmaxf(red[tid], red[tid + st]); __syncthreads(); }
  mx = red[0]; __syncthreads();
  float sum = 0.0f;
  for (int i = tid; i < CLENL; i += 256) { float e = __expf(Smax[b * CLENL + i] - mx); bw[i] = e; sum += e; }
  red[tid] = sum; __syncthreads();
  for (int st = 128; st > 0; st >>= 1) { if (tid < st) red[tid] += red[tid + st]; __syncthreads(); }
  const float inv = 1.0f / red[0];
  __syncthreads();
  const int d = tid;               // 256 threads == H2
  float acc = 0.0f;
  for (int i = 0; i < CLENL; ++i) acc += bw[i] * c[((size_t)b * CLENL + i) * H2 + d];
  ac[(size_t)b * H2 + d] = acc * inv;
}

// ---------------- G = [c, aq, c*aq, c*ac]  (f32 + f16 shadow) ----------------
__global__ __launch_bounds__(256) void build_g(
    const float* __restrict__ c, const float* __restrict__ aq, const float* __restrict__ ac,
    float* __restrict__ Gf, _Float16* __restrict__ Gh) {
  const int m = blockIdx.x;
  const int b = m / CLENL;
  const int d = threadIdx.x;       // 0..255
  const float cv  = c [(size_t)m * H2 + d];
  const float av  = aq[(size_t)m * H2 + d];
  const float acv = ac[(size_t)b * H2 + d];
  const size_t base = (size_t)m * H8;
  const float v0 = cv, v1 = av, v2 = cv * av, v3 = cv * acv;
  Gf[base +          d] = v0;  Gh[base +          d] = (_Float16)v0;
  Gf[base + H2  +    d] = v1;  Gh[base + H2  +    d] = (_Float16)v1;
  Gf[base + 2*H2 +   d] = v2;  Gh[base + 2*H2 +   d] = (_Float16)v2;
  Gf[base + 3*H2 +   d] = v3;  Gh[base + 3*H2 +   d] = (_Float16)v3;
}

// ---------------- logits: [G, M] . W + b  ----------------
__global__ __launch_bounds__(128) void proj_kernel(
    const float* __restrict__ Gf, const float* __restrict__ Mf,
    const float* __restrict__ W, const float* __restrict__ b1, float* __restrict__ out) {
  __shared__ float red[128];
  const int m = blockIdx.x, tid = threadIdx.x;
  float s = 0.0f;
  const float* gr = Gf + (size_t)m * H8;
  for (int d = tid; d < H8; d += 128) s += gr[d] * W[d];
  const float* mr = Mf + (size_t)m * H2;
  for (int d = tid; d < H2; d += 128) s += mr[d] * W[H8 + d];
  red[tid] = s; __syncthreads();
  for (int st = 64; st > 0; st >>= 1) { if (tid < st) red[tid] += red[tid + st]; __syncthreads(); }
  if (tid == 0) out[m] = red[0] + b1[0];
}

// =======================================================================
extern "C" void kernel_launch(void* const* d_in, const int* in_sizes, int n_in,
                              void* d_out, int out_size, void* d_ws, size_t ws_size,
                              hipStream_t stream) {
  // setup_inputs() flat order (nested dicts flattened in insertion order):
  const float* emb     = (const float*)d_in[0];
  const float* wih_cf  = (const float*)d_in[1];
  const float* whh_cf  = (const float*)d_in[2];
  const float* b_cf    = (const float*)d_in[3];
  const float* wih_cb  = (const float*)d_in[4];
  const float* whh_cb  = (const float*)d_in[5];
  const float* b_cb    = (const float*)d_in[6];
  const float* wsim    = (const float*)d_in[7];
  const float* wih_m1f = (const float*)d_in[8];
  const float* whh_m1f = (const float*)d_in[9];
  const float* b_m1f   = (const float*)d_in[10];
  const float* wih_m1b = (const float*)d_in[11];
  const float* whh_m1b = (const float*)d_in[12];
  const float* b_m1b   = (const float*)d_in[13];
  const float* wih_m2f = (const float*)d_in[14];
  const float* whh_m2f = (const float*)d_in[15];
  const float* b_m2f   = (const float*)d_in[16];
  const float* wih_m2b = (const float*)d_in[17];
  const float* whh_m2b = (const float*)d_in[18];
  const float* b_m2b   = (const float*)d_in[19];
  const float* wih_ef  = (const float*)d_in[20];
  const float* whh_ef  = (const float*)d_in[21];
  const float* b_ef    = (const float*)d_in[22];
  const float* wih_eb  = (const float*)d_in[23];
  const float* whh_eb  = (const float*)d_in[24];
  const float* b_eb    = (const float*)d_in[25];
  const float* p1W     = (const float*)d_in[26];
  const float* p1b     = (const float*)d_in[27];
  const float* p2W     = (const float*)d_in[28];
  const float* p2b     = (const float*)d_in[29];
  const int*   ctx     = (const int*)d_in[30];
  const int*   que     = (const int*)d_in[31];
  float* out = (float*)d_out;

  // ---- carve workspace (~115 MB total) ----
  char* base = (char*)d_ws;
  size_t off = 0;
  auto carve = [&](size_t bytes) -> void* {
    void* p = base + off;
    off += (bytes + 255) & ~(size_t)255;
    return p;
  };
  _Float16* wihcf_h  = (_Float16*)carve((size_t)H4 * EMBP * 2);
  _Float16* wihcb_h  = (_Float16*)carve((size_t)H4 * EMBP * 2);
  _Float16* whhcf_h  = (_Float16*)carve((size_t)H4 * HD * 2);
  _Float16* whhcb_h  = (_Float16*)carve((size_t)H4 * HD * 2);
  _Float16* whhm1f_h = (_Float16*)carve((size_t)H4 * HD * 2);
  _Float16* whhm1b_h = (_Float16*)carve((size_t)H4 * HD * 2);
  _Float16* whhm2f_h = (_Float16*)carve((size_t)H4 * HD * 2);
  _Float16* whhm2b_h = (_Float16*)carve((size_t)H4 * HD * 2);
  _Float16* whhef_h  = (_Float16*)carve((size_t)H4 * HD * 2);
  _Float16* whheb_h  = (_Float16*)carve((size_t)H4 * HD * 2);
  _Float16* wihm1f_h = (_Float16*)carve((size_t)H4 * H8 * 2);
  _Float16* wihm1b_h = (_Float16*)carve((size_t)H4 * H8 * 2);
  _Float16* wihm2f_h = (_Float16*)carve((size_t)H4 * H2 * 2);
  _Float16* wihm2b_h = (_Float16*)carve((size_t)H4 * H2 * 2);
  _Float16* wihef_h  = (_Float16*)carve((size_t)H4 * H2 * 2);
  _Float16* wiheb_h  = (_Float16*)carve((size_t)H4 * H2 * 2);
  _Float16* ce_h     = (_Float16*)carve((size_t)MC * EMBP * 2);
  _Float16* qe_h     = (_Float16*)carve((size_t)MQ * EMBP * 2);
  float*    xg0      = (float*)carve((size_t)MC * H4 * 4);
  float*    xg1      = (float*)carve((size_t)MC * H4 * 4);
  float*    cOut     = (float*)carve((size_t)MC * H2 * 4);
  float*    qOut     = (float*)carve((size_t)MQ * H2 * 4);
  float*    cdotB    = (float*)carve((size_t)MC * 4);
  float*    qdotB    = (float*)carve((size_t)MQ * 4);
  float*    SmaxB    = (float*)carve((size_t)MC * 4);
  float*    aqB      = (float*)carve((size_t)MC * H2 * 4);
  float*    acB      = (float*)carve((size_t)BSZ * H2 * 4);
  float*    Gf       = (float*)carve((size_t)MC * H8 * 4);
  _Float16* Gh       = (_Float16*)carve((size_t)MC * H8 * 2);
  float*    M1f      = (float*)carve((size_t)MC * H2 * 4);
  _Float16* M1h      = (_Float16*)carve((size_t)MC * H2 * 2);
  float*    Mf       = (float*)carve((size_t)MC * H2 * 4);
  _Float16* Mh       = (_Float16*)carve((size_t)MC * H2 * 2);
  float*    M2f      = (float*)carve((size_t)MC * H2 * 4);
  (void)ws_size; (void)in_sizes; (void)n_in; (void)out_size;

  auto cdiv = [](int a, int b) { return (a + b - 1) / b; };
#define CVT(src, dst, rows, k, kp) \
  cvt_pad<<<dim3(cdiv((rows) * (kp), 256)), dim3(256), 0, stream>>>(src, dst, rows, k, kp)

  // ---- weight conversion to f16 (padded K) ----
  CVT(wih_cf,  wihcf_h,  H4, EMBD, EMBP);
  CVT(wih_cb,  wihcb_h,  H4, EMBD, EMBP);
  CVT(whh_cf,  whhcf_h,  H4, HD, HD);
  CVT(whh_cb,  whhcb_h,  H4, HD, HD);
  CVT(whh_m1f, whhm1f_h, H4, HD, HD);
  CVT(whh_m1b, whhm1b_h, H4, HD, HD);
  CVT(whh_m2f, whhm2f_h, H4, HD, HD);
  CVT(whh_m2b, whhm2b_h, H4, HD, HD);
  CVT(whh_ef,  whhef_h,  H4, HD, HD);
  CVT(whh_eb,  whheb_h,  H4, HD, HD);
  CVT(wih_m1f, wihm1f_h, H4, H8, H8);
  CVT(wih_m1b, wihm1b_h, H4, H8, H8);
  CVT(wih_m2f, wihm2f_h, H4, H2, H2);
  CVT(wih_m2b, wihm2b_h, H4, H2, H2);
  CVT(wih_ef,  wihef_h,  H4, H2, H2);
  CVT(wih_eb,  wiheb_h,  H4, H2, H2);

  // ---- embeddings ----
  embed_kernel<<<dim3(cdiv(MC * EMBP, 256)), dim3(256), 0, stream>>>(emb, ctx, ce_h, MC);
  embed_kernel<<<dim3(cdiv(MQ * EMBP, 256)), dim3(256), 0, stream>>>(emb, que, qe_h, MQ);

  const dim3 gC(H4 / 64, MC / 16);   // (8, 400)
  const dim3 gQ(H4 / 64, MQ / 16);   // (8, 48)

  // ---- ctx BiLSTM on context ----
  wmma_gemm_bias<<<gC, 32, 0, stream>>>(ce_h, wihcf_h, b_cf, xg0, MC, H4, EMBP);
  wmma_gemm_bias<<<gC, 32, 0, stream>>>(ce_h, wihcb_h, b_cb, xg1, MC, H4, EMBP);
  lstm_recur<<<2, 256, LDS_TOTAL, stream>>>(xg0, xg1, whhcf_h, whhcb_h, cOut, (_Float16*)nullptr, CLENL);

  // ---- ctx BiLSTM on question (same weights) ----
  wmma_gemm_bias<<<gQ, 32, 0, stream>>>(qe_h, wihcf_h, b_cf, xg0, MQ, H4, EMBP);
  wmma_gemm_bias<<<gQ, 32, 0, stream>>>(qe_h, wihcb_h, b_cb, xg1, MQ, H4, EMBP);
  lstm_recur<<<2, 256, LDS_TOTAL, stream>>>(xg0, xg1, whhcf_h, whhcb_h, qOut, (_Float16*)nullptr, QLENL);

  // ---- trilinear attention ----
  dotw_kernel<<<dim3(cdiv(MC, 256)), dim3(256), 0, stream>>>(cOut, wsim,      cdotB, MC);
  dotw_kernel<<<dim3(cdiv(MQ, 256)), dim3(256), 0, stream>>>(qOut, wsim + H2, qdotB, MQ);
  attn_kernel<<<dim3(MC), dim3(64), 0, stream>>>(cOut, qOut, cdotB, qdotB, wsim + 2 * H2, aqB, SmaxB);
  q2c_kernel<<<dim3(BSZ), dim3(256), 0, stream>>>(SmaxB, cOut, acB);
  build_g<<<dim3(MC), dim3(256), 0, stream>>>(cOut, aqB, acB, Gf, Gh);

  // ---- modeling layer 1 (input 8H) ----
  wmma_gemm_bias<<<gC, 32, 0, stream>>>(Gh, wihm1f_h, b_m1f, xg0, MC, H4, H8);
  wmma_gemm_bias<<<gC, 32, 0, stream>>>(Gh, wihm1b_h, b_m1b, xg1, MC, H4, H8);
  lstm_recur<<<2, 256, LDS_TOTAL, stream>>>(xg0, xg1, whhm1f_h, whhm1b_h, M1f, M1h, CLENL);

  // ---- modeling layer 2 (input 2H) -> M ----
  wmma_gemm_bias<<<gC, 32, 0, stream>>>(M1h, wihm2f_h, b_m2f, xg0, MC, H4, H2);
  wmma_gemm_bias<<<gC, 32, 0, stream>>>(M1h, wihm2b_h, b_m2b, xg1, MC, H4, H2);
  lstm_recur<<<2, 256, LDS_TOTAL, stream>>>(xg0, xg1, whhm2f_h, whhm2b_h, Mf, Mh, CLENL);

  // ---- start logits ----
  proj_kernel<<<dim3(MC), dim3(128), 0, stream>>>(Gf, Mf, p1W, p1b, out);

  // ---- end BiLSTM (input 2H) -> M2, end logits ----
  wmma_gemm_bias<<<gC, 32, 0, stream>>>(Mh, wihef_h, b_ef, xg0, MC, H4, H2);
  wmma_gemm_bias<<<gC, 32, 0, stream>>>(Mh, wiheb_h, b_eb, xg1, MC, H4, H2);
  lstm_recur<<<2, 256, LDS_TOTAL, stream>>>(xg0, xg1, whhef_h, whheb_h, M2f, (_Float16*)nullptr, CLENL);
  proj_kernel<<<dim3(MC), dim3(128), 0, stream>>>(Gf, M2f, p2W, p2b, out + MC);
#undef CVT
}
